// _XPermPredictor_2035814498916
// MI455X (gfx1250) — matrix operand order repaired and checked by reading, not compiled, not run on hardware
//
#include <hip/hip_runtime.h>
#include <hip/hip_bf16.h>
#include <math.h>

// ---------------------------------------------------------------------------
// Problem constants (from reference)
// ---------------------------------------------------------------------------
#define NROWS   8192        // B*S
#define HID     4096        // HIDDEN
#define GH      128         // GATE_HIDDEN
#define NCLUST  4           // NUM_CLUSTERS
#define NBUCK   256         // NUM_BUCKETS
#define KITERS  5
#define INNER   (128*32*32) // NUM_BLOCKS*BLOCK_SIZE*BLOCK_SIZE = 131072
#define OUT0    (NBUCK*INNER)

typedef __attribute__((ext_vector_type(16))) __bf16 v16bf;
typedef __attribute__((ext_vector_type(8)))  __bf16 v8bf;
typedef __attribute__((ext_vector_type(8)))  float  v8f;

__device__ __forceinline__ unsigned short f2bfu(float f) {
    unsigned int u = __builtin_bit_cast(unsigned int, f);
    unsigned int r = u + 0x7FFFu + ((u >> 16) & 1u);   // round-to-nearest-even
    return (unsigned short)(r >> 16);
}

// Async global->LDS copy of 16 bytes (CDNA5, ASYNCcnt-tracked).
// LDS destination is the low 32 bits of the shared-aperture flat address.
__device__ __forceinline__ void async_copy_b128(const void* gptr, void* lptr) {
    unsigned lds_off = (unsigned)(size_t)lptr;
    asm volatile("global_load_async_to_lds_b128 %0, %1, off"
                 :: "v"(lds_off), "v"(gptr) : "memory");
}
#define WAIT_ASYNC(N) asm volatile("s_wait_asynccnt " #N ::: "memory")

// ---------------------------------------------------------------------------
// 1) W1 (HID x GH, row-major fp32) -> bf16 transposed w1t (GH x HID) so that
//    B fragments become contiguous-K loads. One-time 2MB pass.
// ---------------------------------------------------------------------------
__global__ void __launch_bounds__(256)
transpose_w1_bf16(const float* __restrict__ W1, unsigned short* __restrict__ w1t, int n) {
    int i = blockIdx.x * blockDim.x + threadIdx.x;
    if (i >= n) return;
    int nn = i / HID;     // 0..127
    int kk = i % HID;     // 0..4095
    w1t[i] = f2bfu(W1[kk * GH + nn]);
}

// ---------------------------------------------------------------------------
// 2) h = gelu(x @ W1 + b1) via v_wmma_f32_16x16x32_bf16.
//    Block = 256 threads = 8 waves. Block tile: 64 (M) x 128 (N).
//    Wave w owns N-tile w (16 cols) and accumulates 4 M-subtiles (64x16):
//    each B fragment is reused by 4 WMMAs.
//    fp32 x tiles (64x32 = 8KB) are async-copied to LDS double-buffered;
//    the fp32->bf16 conversion is fused into fragment assembly (VALU cvt
//    co-executes with XDL WMMA). Even/odd K-steps are manually unrolled so
//    buffer selection and LDS offsets are compile-time constants and the
//    accumulators never move registers at the loop backedge.
// ---------------------------------------------------------------------------
__global__ void __launch_bounds__(256)
gemm1_wmma_gelu(const float*  __restrict__ x,     // [NROWS][HID] fp32 row-major
                const __bf16* __restrict__ w1t,   // [GH][HID]   bf16 row-major (W1^T)
                const float*  __restrict__ b1,    // [GH]
                float*        __restrict__ h)     // [NROWS][GH]
{
    __shared__ float sA[2][64][32];               // double-buffered fp32 A tile, 2 x 8KB

    const int tid  = threadIdx.x;
    const int lane = tid & 31;
    const int wave = tid >> 5;                    // 0..7 -> N tile
    const int m0   = blockIdx.x * 64;
    const int half = (lane >> 4) & 1;             // lane-half select per ISA layout
    const int mn   = lane & 15;

    // staging: each of 256 threads async-copies 32B (2 x b128) per K-step
    const int srow = tid >> 2;                    // 0..63
    const int sseg = (tid & 3) * 8;               // f32 offset 0/8/16/24
    const float* gsrc = x + (size_t)(m0 + srow) * HID + sseg;

    const __bf16* bbase = w1t + (size_t)(wave * 16 + mn) * HID + half * 8;

    v8f acc[4] = {};
    const int NSTEP = HID / 32;                   // 128 (even)

#define STAGE(BUF, STEP) do {                                               \
        const float* g_ = gsrc + (STEP) * 32;                               \
        async_copy_b128(g_,     &sA[BUF][srow][sseg]);                      \
        async_copy_b128(g_ + 4, &sA[BUF][srow][sseg + 4]);                  \
    } while (0)

#define COMPUTE(BUF, STEP) do {                                             \
        const __bf16* bcol_ = bbase + (STEP) * 32;                          \
        v8bf blo_ = *(const v8bf*)(bcol_);                                  \
        v8bf bhi_ = *(const v8bf*)(bcol_ + 16);                             \
        __builtin_prefetch((const void*)(bcol_ + 32), 0, 3);                \
        v16bf b_;                                                           \
        _Pragma("unroll")                                                   \
        for (int e = 0; e < 8; ++e) { b_[e] = blo_[e]; b_[e + 8] = bhi_[e]; } \
        _Pragma("unroll")                                                   \
        for (int s = 0; s < 4; ++s) {                                       \
            const float* arow_ = &sA[BUF][s * 16 + mn][0];                  \
            v8f flo_ = *(const v8f*)(arow_ + half * 8);                     \
            v8f fhi_ = *(const v8f*)(arow_ + 16 + half * 8);                \
            v16bf a_;                                                       \
            _Pragma("unroll")                                               \
            for (int e = 0; e < 8; ++e) {                                   \
                a_[e]     = (__bf16)flo_[e];                                \
                a_[e + 8] = (__bf16)fhi_[e];                                \
            }                                                               \
            acc[s] = __builtin_amdgcn_wmma_f32_16x16x32_bf16(               \
                         false, a_, false, b_, (short)0, acc[s], false, false); \
        }                                                                   \
    } while (0)

    // prologue: stage K-step 0 into buffer 0
    STAGE(0, 0);

    for (int i = 0; i < NSTEP; i += 2) {
        // even step (buffer 0)
        STAGE(1, i + 1);                          // overlap with compute of buf 0
        WAIT_ASYNC(0x2);                          // buf0 pair retired (in-order)
        __syncthreads();
        COMPUTE(0, i);
        __syncthreads();                          // buf0 free for restage

        // odd step (buffer 1)
        if (i + 2 < NSTEP) { STAGE(0, i + 2); WAIT_ASYNC(0x2); }
        else               { WAIT_ASYNC(0x0); }
        __syncthreads();
        COMPUTE(1, i + 1);
        __syncthreads();                          // buf1 free for restage
    }
#undef STAGE
#undef COMPUTE

    // ---- epilogue: bias + exact GELU, write per documented C/D layout ----
    const int   nglob = wave * 16 + mn;
    const float bias  = b1[nglob];
#pragma unroll
    for (int s = 0; s < 4; ++s) {
#pragma unroll
        for (int r = 0; r < 8; ++r) {
            int   m = m0 + s * 16 + r + half * 8;
            float v = acc[s][r] + bias;
            float g = 0.5f * v * (1.0f + erff(v * 0.70710678118654752f));
            h[(size_t)m * GH + nglob] = g;
        }
    }
}

// ---------------------------------------------------------------------------
// 3) gate = softmax(h @ W2 + b2)   (K=128, N=4 -> one row per thread)
// ---------------------------------------------------------------------------
__global__ void __launch_bounds__(256)
gate_softmax(const float* __restrict__ h, const float* __restrict__ W2,
             const float* __restrict__ b2, float* __restrict__ gate, int n) {
    int row = blockIdx.x * blockDim.x + threadIdx.x;
    if (row >= n) return;
    const float* hr = h + (size_t)row * GH;
    float a0 = b2[0], a1 = b2[1], a2 = b2[2], a3 = b2[3];
#pragma unroll 4
    for (int j = 0; j < GH; ++j) {
        float hv = hr[j];
        a0 += hv * W2[j * 4 + 0];
        a1 += hv * W2[j * 4 + 1];
        a2 += hv * W2[j * 4 + 2];
        a3 += hv * W2[j * 4 + 3];
    }
    float mx = fmaxf(fmaxf(a0, a1), fmaxf(a2, a3));
    float e0 = expf(a0 - mx), e1 = expf(a1 - mx), e2 = expf(a2 - mx), e3 = expf(a3 - mx);
    float inv = 1.0f / (e0 + e1 + e2 + e3);
    gate[row * 4 + 0] = e0 * inv;
    gate[row * 4 + 1] = e1 * inv;
    gate[row * 4 + 2] = e2 * inv;
    gate[row * 4 + 3] = e3 * inv;
}

// ---------------------------------------------------------------------------
// k-means helpers
// ---------------------------------------------------------------------------
__global__ void zero_f32(float* __restrict__ p, int n) {
    int i = blockIdx.x * blockDim.x + threadIdx.x;
    if (i < n) p[i] = 0.0f;
}

__global__ void init_centers(const float* __restrict__ gate, float* __restrict__ centers) {
    int t = threadIdx.x;            // 256 threads
#pragma unroll
    for (int d = 0; d < 4; ++d)
        centers[t * 4 + d] = gate[(size_t)(t * (NROWS / NBUCK)) * 4 + d];
}

__global__ void __launch_bounds__(256)
kmeans_assign(const float* __restrict__ gate, const float* __restrict__ centers,
              float* __restrict__ sums, float* __restrict__ cnts,
              int* __restrict__ ids, int n) {
    __shared__ float sc[NBUCK * 4];
    for (int i = threadIdx.x; i < NBUCK * 4; i += 256) sc[i] = centers[i];
    __syncthreads();

    int row = blockIdx.x * blockDim.x + threadIdx.x;
    if (row >= n) return;
    float g0 = gate[row * 4 + 0], g1 = gate[row * 4 + 1];
    float g2 = gate[row * 4 + 2], g3 = gate[row * 4 + 3];
    int best = 0; float bd = 3.4e38f;
#pragma unroll 4
    for (int c = 0; c < NBUCK; ++c) {
        float d0 = g0 - sc[c * 4 + 0];
        float d1 = g1 - sc[c * 4 + 1];
        float d2 = g2 - sc[c * 4 + 2];
        float d3 = g3 - sc[c * 4 + 3];
        float d  = d0 * d0 + d1 * d1 + d2 * d2 + d3 * d3;
        if (d < bd) { bd = d; best = c; }
    }
    ids[row] = best;
    atomicAdd(&sums[best * 4 + 0], g0);
    atomicAdd(&sums[best * 4 + 1], g1);
    atomicAdd(&sums[best * 4 + 2], g2);
    atomicAdd(&sums[best * 4 + 3], g3);
    atomicAdd(&cnts[best], 1.0f);
}

__global__ void kmeans_update(const float* __restrict__ sums, const float* __restrict__ cnts,
                              float* __restrict__ centers, const float* __restrict__ gate) {
    int t = threadIdx.x;            // 256 threads
    float c = cnts[t];
#pragma unroll
    for (int d = 0; d < 4; ++d) {
        float v = (c > 0.0f) ? sums[t * 4 + d] / c
                             : gate[(size_t)((t * 131 + 17) & (NROWS - 1)) * 4 + d];
        centers[t * 4 + d] = v;
    }
}

// ---------------------------------------------------------------------------
// bucketed mean of gate by ids
// ---------------------------------------------------------------------------
__global__ void __launch_bounds__(256)
bucket_accum(const float* __restrict__ gate, const int* __restrict__ ids,
             float* __restrict__ bsum, float* __restrict__ bcnt, int n) {
    int row = blockIdx.x * blockDim.x + threadIdx.x;
    if (row >= n) return;
    int m = ids[row];
    atomicAdd(&bsum[m * 4 + 0], gate[row * 4 + 0]);
    atomicAdd(&bsum[m * 4 + 1], gate[row * 4 + 1]);
    atomicAdd(&bsum[m * 4 + 2], gate[row * 4 + 2]);
    atomicAdd(&bsum[m * 4 + 3], gate[row * 4 + 3]);
    atomicAdd(&bcnt[m], 1.0f);
}

__global__ void bucket_final(const float* __restrict__ bsum, const float* __restrict__ bcnt,
                             float* __restrict__ bg) {
    int t = threadIdx.x;            // 256 threads
    float c = fmaxf(bcnt[t], 1.0f);
#pragma unroll
    for (int d = 0; d < 4; ++d) bg[t * 4 + d] = bsum[t * 4 + d] / c;
}

// ---------------------------------------------------------------------------
// 4) bucket_logits[m, :] = sum_k bg[m,k] * cluster_logits[k, :]
//    K=4, pure bandwidth: float4 streaming, cluster_logits (2MB) L2-resident.
// ---------------------------------------------------------------------------
__global__ void __launch_bounds__(256)
bucket_einsum(const float* __restrict__ bg, const float* __restrict__ cl,
              float* __restrict__ out) {
    int m   = blockIdx.y;                                   // 0..255
    int idx = blockIdx.x * blockDim.x + threadIdx.x;        // over INNER/4
    const float4* cl4 = (const float4*)cl;
    float g0 = bg[m * 4 + 0], g1 = bg[m * 4 + 1];
    float g2 = bg[m * 4 + 2], g3 = bg[m * 4 + 3];
    float4 a = cl4[0 * (INNER / 4) + idx];
    float4 b = cl4[1 * (INNER / 4) + idx];
    float4 c = cl4[2 * (INNER / 4) + idx];
    float4 d = cl4[3 * (INNER / 4) + idx];
    float4 o;
    o.x = g0 * a.x + g1 * b.x + g2 * c.x + g3 * d.x;
    o.y = g0 * a.y + g1 * b.y + g2 * c.y + g3 * d.y;
    o.z = g0 * a.z + g1 * b.z + g2 * c.z + g3 * d.z;
    o.w = g0 * a.w + g1 * b.w + g2 * c.w + g3 * d.w;
    ((float4*)out)[(size_t)m * (INNER / 4) + idx] = o;
}

// ---------------------------------------------------------------------------
// launch
// ---------------------------------------------------------------------------
extern "C" void kernel_launch(void* const* d_in, const int* in_sizes, int n_in,
                              void* d_out, int out_size, void* d_ws, size_t ws_size,
                              hipStream_t stream) {
    const float* tensor = (const float*)d_in[0];   // [4,2048,4096]
    const float* W1     = (const float*)d_in[1];   // [4096,128]
    const float* b1     = (const float*)d_in[2];   // [128]
    const float* W2     = (const float*)d_in[3];   // [128,4]
    const float* b2     = (const float*)d_in[4];   // [4]
    const float* cl     = (const float*)d_in[5];   // [4,128,32,32]
    (void)in_sizes; (void)n_in; (void)ws_size;

    // workspace carve-out
    char* ws = (char*)d_ws;
    unsigned short* w1t_u = (unsigned short*)ws;                         // 1 MB
    float* h       = (float*)(ws + (size_t)(2 << 20));                   // 4 MB
    float* gate    = (float*)(ws + (size_t)(6 << 20));                   // 128 KB
    float* centers = (float*)(ws + (size_t)(6 << 20) + (1 << 17));       // 4 KB
    float* sums    = centers + NBUCK * 4;                                // 4 KB
    float* cnts    = sums + NBUCK * 4;                                   // 1 KB
    float* bsum    = cnts + NBUCK;                                       // 4 KB
    float* bcnt    = bsum + NBUCK * 4;                                   // 1 KB
    float* bg      = bcnt + NBUCK;                                       // 4 KB

    float* bucket_logits = (float*)d_out;
    int*   ids           = (int*)((float*)d_out + OUT0);

    // 1) W1 -> bf16 transpose (one-time, tiny)
    transpose_w1_bf16<<<(GH * HID + 255) / 256, 256, 0, stream>>>(
        W1, w1t_u, GH * HID);

    // 2) WMMA GEMM + fused fp32->bf16 + GELU (64x128 tile, async dbl-buffer)
    gemm1_wmma_gelu<<<NROWS / 64, 256, 0, stream>>>(
        tensor, (const __bf16*)w1t_u, b1, h);

    // 3) gate softmax
    gate_softmax<<<NROWS / 256, 256, 0, stream>>>(h, W2, b2, gate, NROWS);

    // 4) k-means (5 iterations; ids from final assign)
    init_centers<<<1, 256, 0, stream>>>(gate, centers);
    for (int it = 0; it < KITERS; ++it) {
        zero_f32<<<(NBUCK * 5 + 255) / 256, 256, 0, stream>>>(sums, NBUCK * 5); // sums+cnts
        kmeans_assign<<<NROWS / 256, 256, 0, stream>>>(gate, centers, sums, cnts, ids, NROWS);
        kmeans_update<<<1, 256, 0, stream>>>(sums, cnts, centers, gate);
    }

    // 5) bucketed gate mean
    zero_f32<<<(NBUCK * 5 + 255) / 256, 256, 0, stream>>>(bsum, NBUCK * 5);     // bsum+bcnt
    bucket_accum<<<NROWS / 256, 256, 0, stream>>>(gate, ids, bsum, bcnt, NROWS);
    bucket_final<<<1, 256, 0, stream>>>(bsum, bcnt, bg);

    // 6) einsum -> bucket_logits
    dim3 eg(INNER / 4 / 256, NBUCK);
    bucket_einsum<<<eg, 256, 0, stream>>>(bg, cl, bucket_logits);
}
